// GraphNN_76854144794902
// MI455X (gfx1250) — matrix operand order repaired and checked by reading, not compiled, not run on hardware
//
#include <hip/hip_runtime.h>
#include <hip/hip_bf16.h>
#include <cstdint>

#define DEVINL __device__ __forceinline__

typedef __attribute__((ext_vector_type(16))) _Float16 v16h;
typedef __attribute__((ext_vector_type(8)))  float    v8f;

constexpr int B = 8, N = 512, E = 1024, H = 128, T = 8, HOPS = 3;
constexpr int CAP  = E + 16 * T;   // per-graph edge slots after per-type pad-to-16 (1152)
constexpr int CAPT = CAP / 16;     // 72 tiles of 16 edges

// ---------------- WMMA fragment helpers (16x16x32 f16, contiguous-K order for A and B) ----------------

// A-matrix 16x32 f16: lane row = lane&15; lanes 0-15 hold K {0..15}, lanes 16-31 hold
// K {16..31}. Elements 0..15 are one contiguous-K 32B run.
DEVINL v16h load_a16(const _Float16* A, int ld, int k0, int lane) {
  int row = lane & 15;
  int kb  = (lane >> 4) << 4;  // 0 or 16
  const _Float16* p = A + row * ld + k0 + kb;
  union { v16h v; unsigned long long d[4]; } u;
  const unsigned long long* q0 = (const unsigned long long*)p;
  const unsigned long long* q1 = (const unsigned long long*)(p + 8);
  u.d[0] = q0[0]; u.d[1] = q0[1]; u.d[2] = q1[0]; u.d[3] = q1[1];
  return u.v;
}

// B-matrix 32x16 f16: lane holds column n0+(lane&15), 16 contiguous K starting at
// k0 + 16*(lane>=16). Wn stored [n][k] row-major -> one 32B contiguous load.
DEVINL v16h load_b16(const _Float16* Wn, int ld, int n0, int k0, int lane) {
  const _Float16* p = Wn + (size_t)(n0 + (lane & 15)) * ld + k0 + ((lane >> 4) << 4);
  union { v16h v; unsigned long long d[4]; } u;
  const unsigned long long* q = (const unsigned long long*)p;
  u.d[0] = q[0]; u.d[1] = q[1]; u.d[2] = q[2]; u.d[3] = q[3];
  return u.v;
}

DEVINL v8f wmma32(v16h a, v16h b, v8f c) {
  return __builtin_amdgcn_wmma_f32_16x16x32_f16(false, a, false, b, (short)0, c, false, false);
}

// CDNA5 async global->LDS DMA (VGLOBAL encoding, tracked by ASYNCcnt): 16B per lane,
// no VGPR round trip. LDS address = low 32 bits of the generic shared pointer.
DEVINL void async_g2l_b128(uint32_t lds_addr, const void* gptr) {
  asm volatile("global_load_async_to_lds_b128 %0, %1, off"
               :: "v"(lds_addr), "v"(gptr) : "memory");
}
DEVINL void wait_async0() {
  asm volatile("s_wait_asynccnt 0" ::: "memory");
}

DEVINL float sigmoid_(float x) { return 1.f / (1.f + __expf(-x)); }
DEVINL float tanh_(float x)    { return 2.f / (1.f + __expf(-2.f * x)) - 1.f; }

// ---------------- setup kernels (run every call; deterministic) ----------------

__global__ void cvt_f16_k(const float* __restrict__ s, _Float16* __restrict__ d, int n) {
  int i = blockIdx.x * blockDim.x + threadIdx.x;
  if (i < n) d[i] = (_Float16)s[i];
}

// W[r][c] (KxHc, row-major) -> WT[c][r] f16 ([n][k] layout for B fragments)
__global__ void transpose_f16_k(const float* __restrict__ W, _Float16* __restrict__ WT,
                                int K, int Hc) {
  int i = blockIdx.x * blockDim.x + threadIdx.x;
  if (i < K * Hc) { int r = i / Hc, c = i % Hc; WT[(size_t)c * K + r] = (_Float16)W[i]; }
}

__global__ void zero_small_k(int* cnt, int* cur) {
  int i = threadIdx.x;
  if (i < B * T) { cnt[i] = 0; cur[i] = 0; }
}

// argmax over the dense one-hot incidence rows/cols -> sparse src/tgt; histogram types
__global__ void extract_k(const float* __restrict__ n2e, const float* __restrict__ e2n,
                          const int* __restrict__ evec, int* src, int* tgt, int* cnt) {
  int i = blockIdx.x * blockDim.x + threadIdx.x;
  if (i >= B * E) return;
  int b = i / E, e = i % E;
  const float* rowp = n2e + (size_t)i * N;          // node2edge[b,e,:]
  int s = 0;
  for (int n = 0; n < N; ++n) if (rowp[n] > 0.5f) { s = n; break; }
  const float* colp = e2n + (size_t)b * N * E + e;  // edge2node[b,:,e]
  int t = 0;
  for (int n = 0; n < N; ++n) if (colp[(size_t)n * E] > 0.5f) { t = n; break; }
  src[i] = s; tgt[i] = t;
  atomicAdd(&cnt[b * T + evec[i]], 1);
}

__global__ void offsets_k(const int* __restrict__ cnt, int* offs, int* tileType) {
  int b = threadIdx.x;
  if (b >= B) return;
  int off = 0;
  for (int t = 0; t < T; ++t) {
    offs[b * T + t] = off;
    int nt = (cnt[b * T + t] + 15) >> 4;
    for (int k = 0; k < nt; ++k) tileType[b * CAPT + off / 16 + k] = t;
    off += nt * 16;
  }
  for (int k = off / 16; k < CAPT; ++k) tileType[b * CAPT + k] = 0;
}

__global__ void fill_order_k(int* order, int n) {
  int i = blockIdx.x * blockDim.x + threadIdx.x;
  if (i < n) order[i] = -1;
}

__global__ void place_k(const int* __restrict__ evec, const int* __restrict__ offs,
                        int* cur, int* order, int* posOf) {
  int i = blockIdx.x * blockDim.x + threadIdx.x;
  if (i >= B * E) return;
  int b = i / E, e = i % E;
  int t = evec[i];
  int p = offs[b * T + t] + atomicAdd(&cur[b * T + t], 1);
  order[b * CAP + p] = e;
  posOf[i] = p;
}

// ---------------- per-hop kernels ----------------

__global__ void zero2_k(float* a, float* b, int n) {
  int i = blockIdx.x * blockDim.x + threadIdx.x;
  if (i < n) { a[i] = 0.f; b[i] = 0.f; }
}

// gather node states to (type-compacted) edge slots, f32 -> f16
__global__ void gather_k(const float* __restrict__ hin, const int* __restrict__ order,
                         const int* __restrict__ src, const int* __restrict__ tgt,
                         _Float16* __restrict__ emb_bw, _Float16* __restrict__ emb_fw) {
  int blk = blockIdx.x;                 // b*CAP + p
  int b = blk / CAP;
  int e = order[blk];
  int c0 = threadIdx.x * 2;
  _Float16* ob = emb_bw + (size_t)blk * H + c0;
  _Float16* of = emb_fw + (size_t)blk * H + c0;
  if (e < 0) { ob[0] = ob[1] = (_Float16)0.f; of[0] = of[1] = (_Float16)0.f; return; }
  int s = src[b * E + e], t = tgt[b * E + e];
  const float* hs = hin + ((size_t)b * N + s) * H + c0;
  const float* ht = hin + ((size_t)b * N + t) * H + c0;
  ob[0] = (_Float16)hs[0]; ob[1] = (_Float16)hs[1];
  of[0] = (_Float16)ht[0]; of[1] = (_Float16)ht[1];
}

// per-edge-type matvec as dense WMMA GEMM: new[16e,128] = emb[16e,128] @ W_t^T
__global__ void edge_gemm_k(const _Float16* __restrict__ emb_bw,
                            const _Float16* __restrict__ emb_fw,
                            const _Float16* __restrict__ wedge16,
                            const int* __restrict__ tileType,
                            float* __restrict__ new_bw, float* __restrict__ new_fw) {
  __shared__ __attribute__((aligned(16))) _Float16 As[2][16][136];  // +8 pad: 4-bank rotation
  int blk = blockIdx.x;
  int b = blk / CAPT, tile = blk % CAPT;
  int p0 = tile * 16;
  int t  = tileType[blk];
  int tid = threadIdx.x;
  const _Float16* Wb = wedge16 + (size_t)t * H * H;  // [h_out][d] row-major == B [n][k]
  // warm the tile's 32KB weight matrix into cache (one 128B line per thread)
  __builtin_prefetch(Wb + (size_t)tid * 64, 0, 3);
  // async-stage A tiles for both directions: 2 dirs x 16 rows x 256B = 512 x 16B chunks,
  // direct global->LDS DMA (ASYNCcnt), no VGPR round trip
  for (int c = tid; c < 512; c += 256) {
    int dir = c >> 8, r = (c >> 4) & 15, ch = (c & 15) * 8;  // ch in halves (16B chunks)
    const _Float16* s = (dir ? emb_fw : emb_bw) + ((size_t)(b * CAP + p0 + r)) * H + ch;
    async_g2l_b128((uint32_t)(uintptr_t)&As[dir][r][ch], s);
  }
  wait_async0();
  __syncthreads();
  int wave = tid >> 5, lane = tid & 31;
  int dir = wave >> 2, wsub = wave & 3;  // waves 0-3: bw, 4-7: fw; 2 col-tiles each
  float* outp = (dir ? new_fw : new_bw) + ((size_t)(b * CAP + p0)) * H;
  const _Float16* Ad = &As[dir][0][0];
  for (int ct = wsub; ct < 8; ct += 4) {
    v8f acc = {};
    for (int ks = 0; ks < 4; ++ks)
      acc = wmma32(load_a16(Ad, 136, ks * 32, lane),
                   load_b16(Wb, H, ct * 16, ks * 32, lane), acc);
    int col = ct * 16 + (lane & 15), rb = (lane >> 4) << 3;
    for (int v = 0; v < 8; ++v) outp[(rb + v) * H + col] = acc[v];
  }
}

// scatter-add edge messages to nodes (f32 global atomics; dup targets handled by HW)
__global__ void scatter_k(const float* __restrict__ new_bw, const float* __restrict__ new_fw,
                          const int* __restrict__ posOf, const int* __restrict__ src,
                          const int* __restrict__ tgt, float* agg_bw, float* agg_fw) {
  int blk = blockIdx.x;                 // b*E + e
  int b = blk / E;
  int p = posOf[blk];
  int s = src[blk], t = tgt[blk];
  int c0 = threadIdx.x * 2;
  const float* nb = new_bw + ((size_t)b * CAP + p) * H + c0;
  const float* nf = new_fw + ((size_t)b * CAP + p) * H + c0;
  float* ab = agg_bw + ((size_t)b * N + t) * H + c0;
  float* af = agg_fw + ((size_t)b * N + s) * H + c0;
  atomicAdd(&ab[0], nb[0]); atomicAdd(&ab[1], nb[1]);
  atomicAdd(&af[0], nf[0]); atomicAdd(&af[1], nf[1]);
}

// fused GatedFusion + GRU step for a 16-node tile; 8 waves, one 16-col WMMA tile each
__global__ void fuse_gru_k(const float* __restrict__ hin,
                           const float* __restrict__ agg_fw, const float* __restrict__ agg_bw,
                           const _Float16* __restrict__ WfT, const _Float16* __restrict__ WzT,
                           const _Float16* __restrict__ WrT, const _Float16* __restrict__ WtT,
                           const float* __restrict__ bfv, float* __restrict__ hout) {
  __shared__ __attribute__((aligned(16))) float    fwS[16 * 128];  // fw, then reused as agg
  __shared__ __attribute__((aligned(16))) float    bwS[16 * 128];
  __shared__ __attribute__((aligned(16))) float    hS [16 * 128];
  __shared__ __attribute__((aligned(16))) float    zgS[16 * 128];
  __shared__ __attribute__((aligned(16))) _Float16 rhS[16 * 128];
  __shared__ __attribute__((aligned(16))) _Float16 x4s[16 * 520];  // [fw|bw|fw*bw|fw-bw]
  __shared__ __attribute__((aligned(16))) _Float16 x2s[16 * 264];  // [h|agg] then [r*h|agg]
  int blk = blockIdx.x;
  int b = blk >> 5, nt = blk & 31;     // N/16 == 32 tiles per graph
  size_t base = ((size_t)b * N + nt * 16) * H;
  int tid = threadIdx.x, wave = tid >> 5, lane = tid & 31;

  for (int i = tid; i < 16 * 128; i += 256) {
    int m = i >> 7, c = i & 127;
    float fv = agg_fw[base + m * H + c];
    float bv = agg_bw[base + m * H + c];
    float hv = hin[base + m * H + c];
    fwS[i] = fv; bwS[i] = bv; hS[i] = hv;
    x4s[m * 520 +       c] = (_Float16)fv;
    x4s[m * 520 + 128 + c] = (_Float16)bv;
    x4s[m * 520 + 256 + c] = (_Float16)(fv * bv);
    x4s[m * 520 + 384 + c] = (_Float16)(fv - bv);
    x2s[m * 264 + c] = (_Float16)hv;
  }
  __syncthreads();
  {  // fusion gate: z = sigmoid(x4 @ WfT + bf); agg = (1-z)*fw + z*bw
    int ct = wave;
    v8f acc = {};
    for (int ks = 0; ks < 16; ++ks)
      acc = wmma32(load_a16(x4s, 520, ks * 32, lane),
                   load_b16(WfT, 512, ct * 16, ks * 32, lane), acc);
    int col = ct * 16 + (lane & 15), rb = (lane >> 4) << 3;
    float bc = bfv[col];
    for (int v = 0; v < 8; ++v) {
      int m = rb + v;
      float z = sigmoid_(acc[v] + bc);
      float a = (1.f - z) * fwS[m * 128 + col] + z * bwS[m * 128 + col];
      fwS[m * 128 + col] = a;                       // fwS now holds agg
      x2s[m * 264 + 128 + col] = (_Float16)a;
    }
  }
  __syncthreads();
  {  // GRU z and r gates over x2 = [h, agg]
    int ct = wave;
    v8f az = {}, ar = {};
    for (int ks = 0; ks < 8; ++ks) {
      v16h a = load_a16(x2s, 264, ks * 32, lane);
      az = wmma32(a, load_b16(WzT, 256, ct * 16, ks * 32, lane), az);
      ar = wmma32(a, load_b16(WrT, 256, ct * 16, ks * 32, lane), ar);
    }
    int col = ct * 16 + (lane & 15), rb = (lane >> 4) << 3;
    for (int v = 0; v < 8; ++v) {
      int m = rb + v;
      zgS[m * 128 + col] = sigmoid_(az[v]);
      float r = sigmoid_(ar[v]);
      rhS[m * 128 + col] = (_Float16)(r * hS[m * 128 + col]);
    }
  }
  __syncthreads();
  for (int i = tid; i < 16 * 128; i += 256) {       // x2 first half := r*h
    int m = i >> 7, c = i & 127;
    x2s[m * 264 + c] = rhS[i];
  }
  __syncthreads();
  {  // t = tanh([r*h, agg] @ WtT); h' = (1-z)*h + z*t
    int ct = wave;
    v8f at = {};
    for (int ks = 0; ks < 8; ++ks)
      at = wmma32(load_a16(x2s, 264, ks * 32, lane),
                  load_b16(WtT, 256, ct * 16, ks * 32, lane), at);
    int col = ct * 16 + (lane & 15), rb = (lane >> 4) << 3;
    for (int v = 0; v < 8; ++v) {
      int m = rb + v;
      float t = tanh_(at[v]);
      float zg = zgS[m * 128 + col];
      hout[base + m * H + col] = (1.f - zg) * hS[m * 128 + col] + zg * t;
    }
  }
}

// ---------------- launch ----------------

extern "C" void kernel_launch(void* const* d_in, const int* in_sizes, int n_in,
                              void* d_out, int out_size, void* d_ws, size_t ws_size,
                              hipStream_t stream) {
  (void)in_sizes; (void)n_in; (void)out_size; (void)ws_size;
  const float* node_feature = (const float*)d_in[0];
  const int*   edge_vec     = (const int*)d_in[1];
  const float* node2edge    = (const float*)d_in[2];
  const float* edge2node    = (const float*)d_in[3];
  const float* eweight      = (const float*)d_in[4];
  const float* Wz = (const float*)d_in[5];
  const float* Wr = (const float*)d_in[6];
  const float* Wt = (const float*)d_in[7];
  const float* Wf = (const float*)d_in[8];
  const float* bf = (const float*)d_in[9];

  char* w = (char*)d_ws;
  auto carve = [&](size_t bytes) -> void* {
    void* p = (void*)w; w += (bytes + 255) & ~((size_t)255); return p;
  };
  _Float16* wedge16 = (_Float16*)carve((size_t)T * H * H * 2);
  _Float16* WzT     = (_Float16*)carve((size_t)2 * H * H * 2);
  _Float16* WrT     = (_Float16*)carve((size_t)2 * H * H * 2);
  _Float16* WtT     = (_Float16*)carve((size_t)2 * H * H * 2);
  _Float16* WfT     = (_Float16*)carve((size_t)4 * H * H * 2);
  int* src      = (int*)carve((size_t)B * E * 4);
  int* tgt      = (int*)carve((size_t)B * E * 4);
  int* posOf    = (int*)carve((size_t)B * E * 4);
  int* order    = (int*)carve((size_t)B * CAP * 4);
  int* tileType = (int*)carve((size_t)B * CAPT * 4);
  int* cnt      = (int*)carve((size_t)B * T * 4);
  int* cur      = (int*)carve((size_t)B * T * 4);
  int* offs     = (int*)carve((size_t)B * T * 4);
  _Float16* emb_bw = (_Float16*)carve((size_t)B * CAP * H * 2);
  _Float16* emb_fw = (_Float16*)carve((size_t)B * CAP * H * 2);
  float* new_bw = (float*)carve((size_t)B * CAP * H * 4);
  float* new_fw = (float*)carve((size_t)B * CAP * H * 4);
  float* agg_bw = (float*)carve((size_t)B * N * H * 4);
  float* agg_fw = (float*)carve((size_t)B * N * H * 4);
  float* h_cur  = (float*)carve((size_t)B * N * H * 4);

  // one-time (per call) preprocessing
  cvt_f16_k<<<(T * H * H + 255) / 256, 256, 0, stream>>>(eweight, wedge16, T * H * H);
  transpose_f16_k<<<(2 * H * H + 255) / 256, 256, 0, stream>>>(Wz, WzT, 2 * H, H);
  transpose_f16_k<<<(2 * H * H + 255) / 256, 256, 0, stream>>>(Wr, WrT, 2 * H, H);
  transpose_f16_k<<<(2 * H * H + 255) / 256, 256, 0, stream>>>(Wt, WtT, 2 * H, H);
  transpose_f16_k<<<(4 * H * H + 255) / 256, 256, 0, stream>>>(Wf, WfT, 4 * H, H);
  zero_small_k<<<1, 128, 0, stream>>>(cnt, cur);
  extract_k<<<(B * E + 255) / 256, 256, 0, stream>>>(node2edge, edge2node, edge_vec,
                                                     src, tgt, cnt);
  offsets_k<<<1, B, 0, stream>>>(cnt, offs, tileType);
  fill_order_k<<<(B * CAP + 255) / 256, 256, 0, stream>>>(order, B * CAP);
  place_k<<<(B * E + 255) / 256, 256, 0, stream>>>(edge_vec, offs, cur, order, posOf);

  const float* hin = node_feature;
  for (int hop = 0; hop < HOPS; ++hop) {
    float* hout = (hop == HOPS - 1) ? (float*)d_out : h_cur;
    zero2_k<<<(B * N * H + 255) / 256, 256, 0, stream>>>(agg_bw, agg_fw, B * N * H);
    gather_k<<<B * CAP, 64, 0, stream>>>(hin, order, src, tgt, emb_bw, emb_fw);
    edge_gemm_k<<<B * CAPT, 256, 0, stream>>>(emb_bw, emb_fw, wedge16, tileType,
                                              new_bw, new_fw);
    scatter_k<<<B * E, 64, 0, stream>>>(new_bw, new_fw, posOf, src, tgt, agg_bw, agg_fw);
    fuse_gru_k<<<B * (N / 16), 256, 0, stream>>>(hin, agg_fw, agg_bw,
                                                 WfT, WzT, WrT, WtT, bf, hout);
    hin = hout;
  }
}
